// SchNet_7602092114195
// MI455X (gfx1250) — matrix operand order repaired and checked by reading, not compiled, not run on hardware
//
#include <hip/hip_runtime.h>
#include <hip/hip_bf16.h>
#include <math.h>

typedef _Float16 half_t;
typedef __attribute__((ext_vector_type(16))) _Float16 v16h;
typedef __attribute__((ext_vector_type(8)))  float    v8f;

#define BATCH  8
#define NATOM  1024
#define NNBR   64
#define FDIM   128
#define NGAUSS 25
#define KPAD   32
#define NLAYER 3
#define WPL    (4096 + 4 * 16384)   // f16 elems of transposed weights per layer

#if defined(__has_builtin)
#if __has_builtin(__builtin_amdgcn_tensor_load_to_lds) && \
    __has_builtin(__builtin_amdgcn_s_wait_tensorcnt)
#define USE_TDM 1
#endif
#endif
#ifndef USE_TDM
#define USE_TDM 0
#endif

static __device__ __forceinline__ float ssp(float v) {
  float sp = (v > 20.0f) ? v : log1pf(__expf(v));
  return sp - 0.69314718055994530942f;
}

union F16Frag { v16h v; unsigned u[8]; };

// A matrix (MxK, 16-bit) fragment, wave32: M = lane%16, K interleave at 8 per half.
__device__ __forceinline__ v16h load_fragA(const half_t* base, int stride, int m0,
                                           int lane, int k0) {
  F16Frag f;
  int m = m0 + (lane & 15);
  int h = (lane >> 4) & 1;
  const half_t* row = base + (size_t)m * stride + k0;
#pragma unroll
  for (int v = 0; v < 8; ++v) {
    int k = ((v & 4) ? 16 : 0) + h * 8 + (v & 3) * 2;
    f.u[v] = *(const unsigned*)(row + k);
  }
  return f.v;
}

// B matrix (KxN, 16-bit) fragment from transposed [n][k] storage:
// N = lane%16, K interleave at 8 per half (same as A).
__device__ __forceinline__ v16h load_fragB(const half_t* baseT, int stride, int n0,
                                           int lane, int k0) {
  F16Frag f;
  int n = n0 + (lane & 15);
  int h = (lane >> 4) & 1;
  const half_t* row = baseT + (size_t)n * stride + k0;
#pragma unroll
  for (int v = 0; v < 8; ++v)
    f.u[v] = *(const unsigned*)(row + ((v & 4) ? 16 : 0) + h * 8 + (v & 3) * 2);
  return f.v;
}

__device__ __forceinline__ v8f wmma_f16(v16h a, v16h b, v8f c) {
  return __builtin_amdgcn_wmma_f32_16x16x32_f16(false, a, false, b, (short)0, c,
                                                false, false);
}

#if USE_TDM
// 1-D Tensor Data Mover copy: nelem16 f16 elements, global -> LDS.
// Descriptor per CDNA5 ISA 8.3-8.6: group0 = {flags, lds_addr, global_addr, type},
// group1 packs data_size=2B, tensor_dim0 = tile_dim0 = nelem16 (<= 65535), 1-D tile
// (tile_dim1 = 0), stride = nelem16. Groups 2/3 + extra group zero (dims >2 unused).
// Issue from a single wave; arguments are wave-uniform (land in SGPRs).
// This toolchain exposes the 6-argument builtin form (clang-23 lane).
typedef unsigned u32x4 __attribute__((ext_vector_type(4)));
typedef int      i32x8 __attribute__((ext_vector_type(8)));
typedef int      i32x4 __attribute__((ext_vector_type(4)));

__device__ __forceinline__ void tdm_load_1d(void* lds_dst, const void* gsrc,
                                            unsigned nelem16) {
  unsigned lds = (unsigned)(size_t)lds_dst;           // LDS aperture: low 32 bits
  unsigned long long ga = (unsigned long long)(size_t)gsrc;
  u32x4 g0 = { 1u,                                    // count=1, user descriptor
               lds,                                   // lds_addr [63:32]
               (unsigned)(ga & 0xffffffffu),          // global_addr [95:64]
               (unsigned)((ga >> 32) & 0x01ffffffu)   // global_addr [120:96]
                   | (2u << 30) };                    // type=2 ("image")
  i32x8 g1 = { (int)(1u << 16),                       // data_size=1 -> 2 bytes
               (int)((nelem16 & 0xffffu) << 16),      // tensor_dim0[15:0]
               (int)((nelem16 >> 16) & 0xffffu),      // tensor_dim0[31:16]
               (int)((nelem16 & 0xffffu) << 16),      // tile_dim0
               0,                                     // tile_dim1=0, tile_dim2=0
               (int)nelem16,                          // tensor_dim0_stride[31:0]
               0, 0 };
  i32x4 gz4 = { 0, 0, 0, 0 };
  i32x8 gz8 = { 0, 0, 0, 0, 0, 0, 0, 0 };
  __builtin_amdgcn_tensor_load_to_lds(g0, g1, gz4, gz4, gz8, 0);
}
#endif

// ---------------------------------------------------------------- prep kernels

__global__ __launch_bounds__(256) void prep_geom_kernel(
    const float* __restrict__ positions, const float* __restrict__ cell,
    const float* __restrict__ cell_offset, const int* __restrict__ neighbors,
    const float* __restrict__ mask, half_t* __restrict__ fij16,
    float* __restrict__ Cm) {
  int idx = blockIdx.x * 256 + threadIdx.x;
  if (idx >= BATCH * NATOM * NNBR) return;
  int b = idx >> 16;              // A*N = 65536
  int rem = idx & 65535;
  int a = rem >> 6;
  int nbr = neighbors[idx];
  const float* pi = positions + (size_t)(b * NATOM + a) * 3;
  const float* pj = positions + (size_t)(b * NATOM + nbr) * 3;
  const float* co = cell_offset + (size_t)idx * 3;
  const float* cl = cell + b * 9;
  float vx = pj[0] - pi[0] + co[0] * cl[0] + co[1] * cl[3] + co[2] * cl[6];
  float vy = pj[1] - pi[1] + co[0] * cl[1] + co[1] * cl[4] + co[2] * cl[7];
  float vz = pj[2] - pi[2] + co[0] * cl[2] + co[1] * cl[5] + co[2] * cl[8];
  float m = mask[idx];
  float d2 = vx * vx + vy * vy + vz * vz;
  float r = sqrtf(m > 0.0f ? d2 : 1.0f) * m;
  const float width = 5.0f / 24.0f;
  const float coef = -0.5f / (width * width);
  half_t* o = fij16 + (size_t)idx * KPAD;
#pragma unroll
  for (int g = 0; g < NGAUSS; ++g) {
    float d = r - width * (float)g;
    o[g] = (half_t)__expf(coef * d * d);
  }
#pragma unroll
  for (int g = NGAUSS; g < KPAD; ++g) o[g] = (half_t)0.0f;
  float C = (r < 5.0f) ? 0.5f * (__cosf(r * 0.62831853071795864769f) + 1.0f) : 0.0f;
  Cm[idx] = C * m;
}

__global__ __launch_bounds__(256) void prep_embed_kernel(
    const int* __restrict__ z, const float* __restrict__ embed,
    float* __restrict__ x) {
  int idx = blockIdx.x * 256 + threadIdx.x;
  if (idx >= BATCH * NATOM * FDIM) return;
  int atom = idx >> 7, f = idx & 127;
  x[idx] = embed[(size_t)z[atom] * FDIM + f];
}

// Transpose + f16-convert all weights: dst[n*K + k] = src[k*F + n]
__global__ __launch_bounds__(256) void prep_weights_kernel(
    const float* __restrict__ fw1, const float* __restrict__ fw2,
    const float* __restrict__ in2f_w, const float* __restrict__ f2out_w,
    const float* __restrict__ dense_w, half_t* __restrict__ w16) {
  int l = blockIdx.x / 5, m = blockIdx.x % 5;
  int tid = threadIdx.x;
  half_t* dstL = w16 + (size_t)l * WPL;
  if (m == 0) {
    const float* src = fw1 + (size_t)l * NGAUSS * FDIM;
    for (int i = tid; i < FDIM * KPAD; i += 256) {
      int n = i >> 5, k = i & 31;
      dstL[i] = (k < NGAUSS) ? (half_t)src[k * FDIM + n] : (half_t)0.0f;
    }
  } else {
    const float* src;
    half_t* dst = dstL + 4096 + (size_t)(m - 1) * 16384;
    if (m == 1)      src = in2f_w  + (size_t)l * 16384;
    else if (m == 2) src = fw2     + (size_t)l * 16384;
    else if (m == 3) src = f2out_w + (size_t)l * 16384;
    else             src = dense_w + (size_t)l * 16384;
    for (int i = tid; i < 16384; i += 256) {
      int n = i >> 7, k = i & 127;
      dst[i] = (half_t)src[k * FDIM + n];
    }
  }
}

// ---------------------------------------------------------------- y = x @ in2f

__global__ __launch_bounds__(256) void gemm_y_kernel(
    const float* __restrict__ x, const half_t* __restrict__ wT,
    float* __restrict__ y) {
  __shared__ half_t aL[128 * 128];
  __shared__ half_t bL[128 * 128];
  int tid = threadIdx.x;
  int rows0 = blockIdx.x * 128;
#if USE_TDM
  if (tid < 32) tdm_load_1d(bL, wT, FDIM * FDIM);     // DMA weights while we
  for (int i = tid; i < 128 * 128; i += 256)          // convert the A tile
    aL[i] = (half_t)x[(size_t)(rows0 + (i >> 7)) * FDIM + (i & 127)];
  if (tid < 32) __builtin_amdgcn_s_wait_tensorcnt(0);
#else
  for (int i = tid; i < 128 * 128; i += 256) {
    aL[i] = (half_t)x[(size_t)(rows0 + (i >> 7)) * FDIM + (i & 127)];
    bL[i] = wT[i];
  }
#endif
  __syncthreads();
  int wave = tid >> 5, lane = tid & 31;
  v8f vz = {};
  v8f acc[8];
#pragma unroll
  for (int n = 0; n < 8; ++n) acc[n] = vz;
#pragma unroll
  for (int k0 = 0; k0 < 128; k0 += 32) {
    v16h af = load_fragA(aL, 128, wave * 16, lane, k0);
#pragma unroll
    for (int n = 0; n < 8; ++n) {
      v16h bf = load_fragB(bL, 128, n * 16, lane, k0);
      acc[n] = wmma_f16(af, bf, acc[n]);
    }
  }
  int hi8 = (lane & 16) ? 8 : 0;
  int col0 = lane & 15;
#pragma unroll
  for (int n = 0; n < 8; ++n)
#pragma unroll
    for (int r = 0; r < 8; ++r) {
      int row = rows0 + wave * 16 + r + hi8;
      y[(size_t)row * FDIM + n * 16 + col0] = acc[n][r];
    }
}

// ------------------------------------------- fused filter-net + CFConv per atom

__global__ __launch_bounds__(256) void cfconv_kernel(
    const half_t* __restrict__ fij16, const float* __restrict__ Cm,
    const int* __restrict__ neighbors, const half_t* __restrict__ w1T,
    const half_t* __restrict__ w2T, const float* __restrict__ fb1,
    const float* __restrict__ fb2, const float* __restrict__ y,
    float* __restrict__ agg) {
  __shared__ half_t fijL[NNBR * KPAD];   //  4 KB
  __shared__ half_t w1L[FDIM * KPAD];    //  8 KB
  __shared__ half_t w2L[FDIM * FDIM];    // 32 KB
  __shared__ half_t hL[NNBR * FDIM];     // 16 KB
  __shared__ float  yL[NNBR * FDIM];     // 32 KB
  __shared__ float  CmL[NNBR];
  __shared__ float  b1L[FDIM], b2L[FDIM];
  __shared__ float  aggL[FDIM];
  __shared__ int    nbrL[NNBR];
  int tid = threadIdx.x;
  int p = blockIdx.x;                    // flat atom index in [0, B*A)
  int baseRow = (p >> 10) << 10;         // b * NATOM
#if USE_TDM
  if (tid < 32) {                        // wave 0 drives the TDM; DMA overlaps
    tdm_load_1d(fijL, fij16 + (size_t)p * NNBR * KPAD, NNBR * KPAD);
    tdm_load_1d(w1L, w1T, FDIM * KPAD);
    tdm_load_1d(w2L, w2T, FDIM * FDIM);
  }
#else
  for (int i = tid; i < NNBR * KPAD; i += 256)
    fijL[i] = fij16[(size_t)p * NNBR * KPAD + i];
  for (int i = tid; i < FDIM * KPAD; i += 256) w1L[i] = w1T[i];
  for (int i = tid; i < FDIM * FDIM; i += 256) w2L[i] = w2T[i];
#endif
  if (tid < NNBR) {
    nbrL[tid] = neighbors[(size_t)p * NNBR + tid];
    CmL[tid] = Cm[(size_t)p * NNBR + tid];
  }
  if (tid < FDIM) { b1L[tid] = fb1[tid]; b2L[tid] = fb2[tid]; aggL[tid] = 0.0f; }
  __syncthreads();
  for (int i = tid; i < NNBR * FDIM; i += 256) {      // neighbor-feature gather
    int row = i >> 7, col = i & 127;
    yL[i] = y[(size_t)(baseRow + nbrL[row]) * FDIM + col];
  }
#if USE_TDM
  if (tid < 32) __builtin_amdgcn_s_wait_tensorcnt(0);
#endif
  __syncthreads();

  int wave = tid >> 5, lane = tid & 31;
  int mt = wave & 3, ntb = (wave >> 2) * 4;   // 4 M-tiles x 8 N-tiles over 8 waves
  int hi8 = (lane & 16) ? 8 : 0;
  int col0 = lane & 15;
  v8f vz = {};
  v8f acc[4];

  // GEMM1: H = fij(64x32) @ w1(32x128) — single WMMA K-step
  {
    v16h af = load_fragA(fijL, KPAD, mt * 16, lane, 0);
#pragma unroll
    for (int j = 0; j < 4; ++j) {
      v16h bf = load_fragB(w1L, KPAD, (ntb + j) * 16, lane, 0);
      acc[j] = wmma_f16(af, bf, vz);
    }
  }
#pragma unroll
  for (int j = 0; j < 4; ++j)
#pragma unroll
    for (int r = 0; r < 8; ++r) {
      int row = mt * 16 + r + hi8;
      int col = (ntb + j) * 16 + col0;
      hL[row * FDIM + col] = (half_t)ssp(acc[j][r] + b1L[col]);
    }
  __syncthreads();

  // GEMM2: W = H(64x128) @ w2(128x128)
#pragma unroll
  for (int j = 0; j < 4; ++j) acc[j] = vz;
#pragma unroll
  for (int k0 = 0; k0 < FDIM; k0 += 32) {
    v16h af = load_fragA(hL, FDIM, mt * 16, lane, k0);
#pragma unroll
    for (int j = 0; j < 4; ++j) {
      v16h bf = load_fragB(w2L, FDIM, (ntb + j) * 16, lane, k0);
      acc[j] = wmma_f16(af, bf, acc[j]);
    }
  }

  // epilogue: (W+b2) * cutoff*mask * gathered y, reduce over 64 neighbor rows
#pragma unroll
  for (int j = 0; j < 4; ++j) {
    int col = (ntb + j) * 16 + col0;
    float partial = 0.0f;
#pragma unroll
    for (int r = 0; r < 8; ++r) {
      int row = mt * 16 + r + hi8;
      partial += (acc[j][r] + b2L[col]) * CmL[row] * yL[row * FDIM + col];
    }
    atomicAdd(&aggL[col], partial);   // ds_add_f32
  }
  __syncthreads();
  if (tid < FDIM) agg[(size_t)p * FDIM + tid] = aggL[tid];
}

// ---------------------- v = ssp(agg@f2out + b) @ dense + db ;  x += v

__global__ __launch_bounds__(256) void outnet_kernel(
    const float* __restrict__ agg, const half_t* __restrict__ f2outT,
    const half_t* __restrict__ denseT, const float* __restrict__ f2out_b,
    const float* __restrict__ dense_b, float* __restrict__ x) {
  __shared__ half_t aL[128 * 128];   // A tile, reused for intermediate t
  __shared__ half_t wAL[128 * 128];
  __shared__ half_t wBL[128 * 128];
  __shared__ float  bAL[FDIM], bBL[FDIM];
  int tid = threadIdx.x;
  int rows0 = blockIdx.x * 128;
  if (tid < FDIM) { bAL[tid] = f2out_b[tid]; bBL[tid] = dense_b[tid]; }
#if USE_TDM
  if (tid < 32) {
    tdm_load_1d(wAL, f2outT, FDIM * FDIM);
    tdm_load_1d(wBL, denseT, FDIM * FDIM);
  }
  for (int i = tid; i < 128 * 128; i += 256)
    aL[i] = (half_t)agg[(size_t)(rows0 + (i >> 7)) * FDIM + (i & 127)];
  if (tid < 32) __builtin_amdgcn_s_wait_tensorcnt(0);
#else
  for (int i = tid; i < 128 * 128; i += 256) {
    aL[i]  = (half_t)agg[(size_t)(rows0 + (i >> 7)) * FDIM + (i & 127)];
    wAL[i] = f2outT[i];
    wBL[i] = denseT[i];
  }
#endif
  __syncthreads();
  int wave = tid >> 5, lane = tid & 31;
  int hi8 = (lane & 16) ? 8 : 0;
  int col0 = lane & 15;
  v8f vz = {};
  v8f acc[8];
#pragma unroll
  for (int n = 0; n < 8; ++n) acc[n] = vz;
#pragma unroll
  for (int k0 = 0; k0 < 128; k0 += 32) {
    v16h af = load_fragA(aL, 128, wave * 16, lane, k0);
#pragma unroll
    for (int n = 0; n < 8; ++n) {
      v16h bf = load_fragB(wAL, 128, n * 16, lane, k0);
      acc[n] = wmma_f16(af, bf, acc[n]);
    }
  }
  __syncthreads();                      // all waves done reading aL
#pragma unroll
  for (int n = 0; n < 8; ++n)
#pragma unroll
    for (int r = 0; r < 8; ++r) {
      int row = wave * 16 + r + hi8;
      int col = n * 16 + col0;
      aL[row * FDIM + col] = (half_t)ssp(acc[n][r] + bAL[col]);
    }
  __syncthreads();
#pragma unroll
  for (int n = 0; n < 8; ++n) acc[n] = vz;
#pragma unroll
  for (int k0 = 0; k0 < 128; k0 += 32) {
    v16h af = load_fragA(aL, 128, wave * 16, lane, k0);
#pragma unroll
    for (int n = 0; n < 8; ++n) {
      v16h bf = load_fragB(wBL, 128, n * 16, lane, k0);
      acc[n] = wmma_f16(af, bf, acc[n]);
    }
  }
#pragma unroll
  for (int n = 0; n < 8; ++n)
#pragma unroll
    for (int r = 0; r < 8; ++r) {
      int row = rows0 + wave * 16 + r + hi8;
      int col = n * 16 + col0;
      size_t o = (size_t)row * FDIM + col;
      x[o] += acc[n][r] + bBL[col];
    }
}

// ---------------------------------------------------------------------- launch

extern "C" void kernel_launch(void* const* d_in, const int* in_sizes, int n_in,
                              void* d_out, int out_size, void* d_ws, size_t ws_size,
                              hipStream_t stream) {
  (void)in_sizes; (void)n_in; (void)out_size; (void)ws_size;
  const int*   atomic_numbers = (const int*)d_in[0];
  const float* positions      = (const float*)d_in[1];
  const float* cell           = (const float*)d_in[2];
  const float* cell_offset    = (const float*)d_in[3];
  const int*   neighbors      = (const int*)d_in[4];
  const float* neighbor_mask  = (const float*)d_in[5];
  const float* embed          = (const float*)d_in[6];
  const float* fw1            = (const float*)d_in[7];
  const float* fb1            = (const float*)d_in[8];
  const float* fw2            = (const float*)d_in[9];
  const float* fb2            = (const float*)d_in[10];
  const float* in2f_w         = (const float*)d_in[11];
  const float* f2out_w        = (const float*)d_in[12];
  const float* f2out_b        = (const float*)d_in[13];
  const float* dense_w        = (const float*)d_in[14];
  const float* dense_b        = (const float*)d_in[15];
  float* x = (float*)d_out;

  char* ws = (char*)d_ws;
  half_t* fij16 = (half_t*)ws;
  size_t off = (size_t)BATCH * NATOM * NNBR * KPAD * sizeof(half_t); // 32 MB
  float* Cm = (float*)(ws + off);  off += (size_t)BATCH * NATOM * NNBR * 4;
  float* y  = (float*)(ws + off);  off += (size_t)BATCH * NATOM * FDIM * 4;
  float* ag = (float*)(ws + off);  off += (size_t)BATCH * NATOM * FDIM * 4;
  half_t* w16 = (half_t*)(ws + off);

  prep_geom_kernel<<<BATCH * NATOM * NNBR / 256, 256, 0, stream>>>(
      positions, cell, cell_offset, neighbors, neighbor_mask, fij16, Cm);
  prep_weights_kernel<<<NLAYER * 5, 256, 0, stream>>>(
      fw1, fw2, in2f_w, f2out_w, dense_w, w16);
  prep_embed_kernel<<<BATCH * NATOM * FDIM / 256, 256, 0, stream>>>(
      atomic_numbers, embed, x);

  for (int l = 0; l < NLAYER; ++l) {
    const half_t* Lw = w16 + (size_t)l * WPL;
    const half_t* w1T    = Lw;                       // 128 x 32
    const half_t* in2fT  = Lw + 4096;                // 128 x 128
    const half_t* w2T    = Lw + 4096 + 16384;
    const half_t* f2outT = Lw + 4096 + 2 * 16384;
    const half_t* denseT = Lw + 4096 + 3 * 16384;
    gemm_y_kernel<<<BATCH * NATOM / 128, 256, 0, stream>>>(x, in2fT, y);
    cfconv_kernel<<<BATCH * NATOM, 256, 0, stream>>>(
        fij16, Cm, neighbors, w1T, w2T, fb1 + l * FDIM, fb2 + l * FDIM, y, ag);
    outnet_kernel<<<BATCH * NATOM / 128, 256, 0, stream>>>(
        ag, f2outT, denseT, f2out_b + l * FDIM, dense_b + l * FDIM, x);
  }
}
